// LocalSoundModel_85916525789693
// MI455X (gfx1250) — matrix operand order repaired and checked by reading, not compiled, not run on hardware
//
#include <hip/hip_runtime.h>
#include <hip/hip_bf16.h>

#define N_SAMP 256
#define CDIM   512
#define PIX    196
#define MTILES 13
#define EPSP 0.65f
#define EPSN 0.40f
#define TAUC 0.03f
#define TEMP 0.07f

typedef _Float16 v16h __attribute__((ext_vector_type(16)));
typedef float    v8f  __attribute__((ext_vector_type(8)));
typedef _Float16 h4   __attribute__((ext_vector_type(4)));
typedef int      v4i  __attribute__((ext_vector_type(4)));

union Frag16 { v16h v; uint4 u[2]; };

// ---- CDNA5 async global->LDS copy (async/tensor path), guarded for toolchain ----
#if defined(__gfx1250__) && __has_builtin(__builtin_amdgcn_global_load_async_to_lds_b128)
#define HAS_ASYNC_LDS 1
#else
#define HAS_ASYNC_LDS 0
#endif

__device__ __forceinline__ void async_copy16(const uint4* g, uint4* l) {
#if HAS_ASYNC_LDS
    __builtin_amdgcn_global_load_async_to_lds_b128((v4i*)g, (v4i*)l, 0, 0);
#else
    *l = *g;
#endif
}

__device__ __forceinline__ void async_wait_all() {
#if HAS_ASYNC_LDS
#if __has_builtin(__builtin_amdgcn_s_wait_asynccnt)
    __builtin_amdgcn_s_wait_asynccnt(0);
#else
    asm volatile("s_wait_asynccnt 0" ::: "memory");
#endif
#endif
}

__device__ __forceinline__ float sigm(float x) { return 1.0f / (1.0f + __expf(-x)); }

// ------------------------------------------------------------------
// Kernel 1: normalize audio rows (C=512), write f16 copy to workspace
// ------------------------------------------------------------------
__global__ __launch_bounds__(256)
void audio_norm_kernel(const float* __restrict__ audio, _Float16* __restrict__ out)
{
    __shared__ float red[8];
    const int row = blockIdx.x;
    const int tid = threadIdx.x;
    const float a0 = audio[row * CDIM + tid];
    const float a1 = audio[row * CDIM + 256 + tid];
    float ss = a0 * a0 + a1 * a1;
#pragma unroll
    for (int m = 1; m < 32; m <<= 1) ss += __shfl_xor(ss, m, 32);
    if ((tid & 31) == 0) red[tid >> 5] = ss;
    __syncthreads();
    float tot = 0.f;
#pragma unroll
    for (int i = 0; i < 8; ++i) tot += red[i];
    const float inv = rsqrtf(tot);
    out[row * CDIM + tid]       = (_Float16)(a0 * inv);
    out[row * CDIM + 256 + tid] = (_Float16)(a1 * inv);
}

// ------------------------------------------------------------------
// Kernel 2: per-sample fused GEMM (WMMA f16->f32) + sigmoid epilogue
//   block n: S[k,p] = f_hat[n,p,:] . a_hat[k,:]  for all k, p
//   LDS: B = audio f16 [256][512] (256KB), A tile = 16 pixels [16][512] (16KB)
// ------------------------------------------------------------------
__global__ __launch_bounds__(256, 1)
void lsm_main_kernel(const float* __restrict__ frame,
                     const _Float16* __restrict__ audio_h,
                     float* __restrict__ loss_part)
{
    extern __shared__ char smem[];
    _Float16* ldsB = (_Float16*)smem;                       // [256][512] f16
    _Float16* ldsA = (_Float16*)(smem + 262144);            // [16][512]  f16
    float*    red  = (float*)(smem + 262144 + 16384);       // 64 floats

    const int n    = blockIdx.x;
    const int tid  = threadIdx.x;
    const int wid  = tid >> 5;
    const int lane = tid & 31;

    // ---- stage f16 audio into LDS (async global->LDS on CDNA5) ----
    {
        const uint4* g = (const uint4*)audio_h;
        uint4* l = (uint4*)ldsB;
        for (int i = tid; i < (N_SAMP * CDIM) / 8; i += 256)
            async_copy16(g + i, l + i);
        async_wait_all();
    }
    __syncthreads();

    // persistent per-lane accumulators (Ni_easy num/den for 2 k-columns,
    // plus diagonal Pi / Ni_hard statistics)
    float num0 = 0.f, den0 = 0.f, num1 = 0.f, den1 = 0.f;
    float pin = 0.f, pid = 0.f, nhn = 0.f, nhd = 0.f;

    const int  mrow   = lane & 15;
    const int  aoff   = (lane < 16) ? 0 : 8;    // A frag: K 0..7/16..23 vs 8..15/24..31
    const int  koffB  = (lane < 16) ? 0 : 8;    // B frag: K 0..7/16..23 vs 8..15/24..31
    const int  col0   = wid * 32 + (lane & 15);
    const int  col1   = col0 + 16;
    const bool diagw  = (wid == (n >> 5));
    const int  rowadd = (lane >> 4) << 3;       // C frag: rows 0..7 vs 8..15

    const _Float16* aBase  = ldsA + mrow * CDIM + aoff;
    const _Float16* b0Base = ldsB + col0 * CDIM + koffB;
    const _Float16* b1Base = ldsB + col1 * CDIM + koffB;

    for (int mt = 0; mt < MTILES; ++mt) {
        // ---- Phase A: load + L2-normalize 16 frame pixel rows -> f16 LDS A tile ----
#pragma unroll
        for (int rr = 0; rr < 2; ++rr) {
            const int r = wid * 2 + rr;
            const int p = mt * 16 + r;
            if (p < PIX) {
                const float4* src = (const float4*)(frame + ((size_t)n * PIX + p) * CDIM);
                float4 vals[4];
                float ss = 0.f;
#pragma unroll
                for (int j = 0; j < 4; ++j) {
                    float4 v = src[j * 32 + lane];
                    vals[j] = v;
                    ss += v.x * v.x + v.y * v.y + v.z * v.z + v.w * v.w;
                }
#pragma unroll
                for (int m = 1; m < 32; m <<= 1) ss += __shfl_xor(ss, m, 32);
                const float inv = rsqrtf(ss);
#pragma unroll
                for (int j = 0; j < 4; ++j) {
                    h4 hv;
                    hv.x = (_Float16)(vals[j].x * inv);
                    hv.y = (_Float16)(vals[j].y * inv);
                    hv.z = (_Float16)(vals[j].z * inv);
                    hv.w = (_Float16)(vals[j].w * inv);
                    *(h4*)&ldsA[r * CDIM + (j * 32 + lane) * 4] = hv;
                }
                // prefetch next M-tile's row
                const int pn = p + 16;
                if (pn < PIX)
                    __builtin_prefetch(frame + ((size_t)n * PIX + pn) * CDIM + lane * 16, 0, 1);
            } else {
                h4 z; z.x = z.y = z.z = z.w = (_Float16)0.f;
#pragma unroll
                for (int j = 0; j < 4; ++j)
                    *(h4*)&ldsA[r * CDIM + (j * 32 + lane) * 4] = z;
            }
        }
        __syncthreads();

        // ---- Phase B: (16x512) x (512x32) via 16 K-steps of WMMA f16 ----
        v8f acc0, acc1;
#pragma unroll
        for (int j = 0; j < 8; ++j) { acc0[j] = 0.f; acc1[j] = 0.f; }

#pragma unroll
        for (int ks = 0; ks < 16; ++ks) {
            Frag16 fa, fb0, fb1;
            fa.u[0]  = *(const uint4*)(aBase  + ks * 32);
            fa.u[1]  = *(const uint4*)(aBase  + ks * 32 + 16);
            fb0.u[0] = *(const uint4*)(b0Base + ks * 32);
            fb0.u[1] = *(const uint4*)(b0Base + ks * 32 + 16);
            fb1.u[0] = *(const uint4*)(b1Base + ks * 32);
            fb1.u[1] = *(const uint4*)(b1Base + ks * 32 + 16);
            acc0 = __builtin_amdgcn_wmma_f32_16x16x32_f16(false, fa.v, false, fb0.v,
                                                          (short)0, acc0, false, false);
            acc1 = __builtin_amdgcn_wmma_f32_16x16x32_f16(false, fa.v, false, fb1.v,
                                                          (short)0, acc1, false, false);
        }

        // ---- Phase C: fused sigmoid-weighted pixel reduction ----
#pragma unroll
        for (int j = 0; j < 8; ++j) {
            const int p = mt * 16 + rowadd + j;
            const float valid = (p < PIX) ? 1.f : 0.f;
            const float s0 = acc0[j], s1 = acc1[j];
            const float w0 = valid * sigm((s0 - EPSP) * (1.0f / TAUC));
            const float w1 = valid * sigm((s1 - EPSP) * (1.0f / TAUC));
            num0 += w0 * s0; den0 += w0;
            num1 += w1 * s1; den1 += w1;
            if (diagw) {  // wave-uniform: this wave owns column k == n
                const float d0 = (col0 == n) ? valid : 0.f;
                const float d1 = (col1 == n) ? valid : 0.f;
                pin += d0 * w0 * s0 + d1 * w1 * s1;
                pid += d0 * w0 + d1 * w1;
                const float g0 = 1.f - sigm((s0 - EPSN) * (1.0f / TAUC));
                const float g1 = 1.f - sigm((s1 - EPSN) * (1.0f / TAUC));
                nhn += d0 * g0 * s0 + d1 * g1 * s1;
                nhd += d0 * g0 + d1 * g1;
            }
        }
        __syncthreads();   // protect ldsA before next tile is staged
    }

    // ---- final: combine the two row-halves (lanes l and l+16 share a column) ----
    num0 += __shfl_xor(num0, 16, 32); den0 += __shfl_xor(den0, 16, 32);
    num1 += __shfl_xor(num1, 16, 32); den1 += __shfl_xor(den1, 16, 32);
    pin  += __shfl_xor(pin, 16, 32);  pid  += __shfl_xor(pid, 16, 32);
    nhn  += __shfl_xor(nhn, 16, 32);  nhd  += __shfl_xor(nhd, 16, 32);

    float ni = 0.f;
    if (lane < 16) {
        const float e0 = num0 / den0;
        const float e1 = num1 / den1;
        ni  = e0 * ((col0 == n) ? -99.f : 1.f);   // mask = 1 - 100*eye
        ni += e1 * ((col1 == n) ? -99.f : 1.f);
    } else {
        pin = pid = nhn = nhd = 0.f;
    }
#pragma unroll
    for (int m = 1; m < 16; m <<= 1) {
        ni  += __shfl_xor(ni,  m, 32);
        pin += __shfl_xor(pin, m, 32);
        pid += __shfl_xor(pid, m, 32);
        nhn += __shfl_xor(nhn, m, 32);
        nhd += __shfl_xor(nhd, m, 32);
    }
    if (lane == 0) {
        red[wid]      = ni;
        red[8  + wid] = pin;
        red[16 + wid] = pid;
        red[24 + wid] = nhn;
        red[32 + wid] = nhd;
    }
    __syncthreads();
    if (tid == 0) {
        float niS = 0, pinS = 0, pidS = 0, nhnS = 0, nhdS = 0;
#pragma unroll
        for (int i = 0; i < 8; ++i) {
            niS += red[i]; pinS += red[8 + i]; pidS += red[16 + i];
            nhnS += red[24 + i]; nhdS += red[32 + i];
        }
        const float Pi = pinS / pidS;
        const float Nh = nhnS / nhdS;
        const float Pd = Pi * TEMP;
        const float Nd = (Nh + niS) * TEMP;
        const float x  = Nd - Pd;
        // -log(sigmoid(Pd - Nd)) = softplus(Nd - Pd), numerically stable
        loss_part[n] = fmaxf(x, 0.f) + log1pf(__expf(-fabsf(x)));
    }
}

// ------------------------------------------------------------------
// Kernel 3: reduce 256 per-sample terms -> scalar loss
// ------------------------------------------------------------------
__global__ __launch_bounds__(256)
void lsm_finalize_kernel(const float* __restrict__ loss_part, float* __restrict__ out)
{
    __shared__ float red[8];
    const int tid = threadIdx.x;
    float v = loss_part[tid];
#pragma unroll
    for (int m = 1; m < 32; m <<= 1) v += __shfl_xor(v, m, 32);
    if ((tid & 31) == 0) red[tid >> 5] = v;
    __syncthreads();
    if (tid == 0) {
        float t = 0.f;
        for (int i = 0; i < 8; ++i) t += red[i];
        out[0] = t * (1.0f / (float)N_SAMP);
    }
}

extern "C" void kernel_launch(void* const* d_in, const int* in_sizes, int n_in,
                              void* d_out, int out_size, void* d_ws, size_t ws_size,
                              hipStream_t stream)
{
    (void)in_sizes; (void)n_in; (void)out_size; (void)ws_size;
    const float* frame = (const float*)d_in[0];   // (256,14,14,512) f32
    const float* audio = (const float*)d_in[1];   // (256,512) f32

    _Float16* ws_audio = (_Float16*)d_ws;                                   // 256KB
    float* ws_loss = (float*)((char*)d_ws + (size_t)N_SAMP * CDIM * sizeof(_Float16)); // 1KB

    audio_norm_kernel<<<N_SAMP, 256, 0, stream>>>(audio, ws_audio);

    const size_t smem = 262144 + 16384 + 64 * sizeof(float);  // 272.25 KB dynamic LDS
    lsm_main_kernel<<<N_SAMP, 256, smem, stream>>>(frame, ws_audio, ws_loss);

    lsm_finalize_kernel<<<1, 256, 0, stream>>>(ws_loss, (float*)d_out);
}
